// PairwiseCosine_17128329576706
// MI455X (gfx1250) — compile-verified, matrix operand order checked by reading, not run on hardware
//
#include <hip/hip_runtime.h>

typedef __attribute__((ext_vector_type(16))) __bf16 v16bf;
typedef __attribute__((ext_vector_type(8)))  __bf16 v8bf;
typedef __attribute__((ext_vector_type(8)))  float  v8f;
typedef __attribute__((ext_vector_type(4)))  unsigned short us4;

#define D_DIM    512
#define BM       128
#define BN       128
#define KC       32
#define LDK      40              // KC + 8 ushort pad -> 80B stride, 16B aligned, conflict-free
#define PLANE    (BM * LDK)      // ushorts per LDS plane tile
#define NCHUNK   (D_DIM / KC)
#define NTHREADS 256
#define EPS_F    1e-8f

__device__ __forceinline__ unsigned int f2bf_bits(float f) {
  unsigned int u = __float_as_uint(f);
  return (u + 0x7FFFu + ((u >> 16) & 1u)) >> 16;   // RNE fp32 -> bf16
}
__device__ __forceinline__ float bfbits2f(unsigned int b) {
  return __uint_as_float(b << 16);
}

__device__ __forceinline__ v8f wmma_bf16(v16bf a, v16bf b, v8f c) {
  return __builtin_amdgcn_wmma_f32_16x16x32_bf16(false, a, false, b, (short)0, c,
                                                 false, false);
}

// CDNA5 async global->LDS copy, 16B per lane, tracked by ASYNCcnt.
// VDST operand = VGPR holding LDS byte address (low 32b of generic shared ptr).
__device__ __forceinline__ void async_copy16(const unsigned short* g, unsigned short* l) {
  unsigned int       loff = (unsigned int)(size_t)l;
  unsigned long long ga   = (unsigned long long)(size_t)g;
  asm volatile("global_load_async_to_lds_b128 %0, %1, off"
               :: "v"(loff), "v"(ga) : "memory");
}
__device__ __forceinline__ void wait_async0() {
  asm volatile("s_wait_asynccnt 0x0" ::: "memory");
}

// ---------------------------------------------------------------------------
// Pre-pass: per-row L2 norm + fp32 -> (bf16 hi, bf16 lo) plane split.
// One wave32 per row; pure streaming, hides the 64x-redundant conversion work
// that used to sit inside the GEMM inner loop.
// ---------------------------------------------------------------------------
__global__ void convert_norms_kernel(const float* __restrict__ x,
                                     unsigned short* __restrict__ hi,
                                     unsigned short* __restrict__ lo,
                                     float* __restrict__ nrm, int rows) {
  const int lane = threadIdx.x & 31;
  const int wave = threadIdx.x >> 5;
  const int row  = blockIdx.x * 8 + wave;
  if (row >= rows) return;
  const float* p = x + (size_t)row * D_DIM;
  float s = 0.f;
#pragma unroll
  for (int j = 0; j < D_DIM / 128; ++j) {     // 4 float4 per lane
    const int k = j * 128 + lane * 4;
    const float4 v = *(const float4*)(p + k);
    const float a[4] = {v.x, v.y, v.z, v.w};
    us4 h, l;
#pragma unroll
    for (int c = 0; c < 4; ++c) {
      s += a[c] * a[c];
      const unsigned int hb = f2bf_bits(a[c]);
      h[c] = (unsigned short)hb;
      l[c] = (unsigned short)f2bf_bits(a[c] - bfbits2f(hb));
    }
    *(us4*)(hi + (size_t)row * D_DIM + k) = h;
    *(us4*)(lo + (size_t)row * D_DIM + k) = l;
  }
#pragma unroll
  for (int off = 16; off > 0; off >>= 1) s += __shfl_xor(s, off, 32);
  if (lane == 0) nrm[row] = sqrtf(s);
}

// ---------------------------------------------------------------------------
// Main GEMM on pre-split bf16 planes: 128x128 tile / 256 threads / 8 waves,
// 3xBF16 fp32 emulation, async-LDS double buffering.
// ---------------------------------------------------------------------------
__global__ __launch_bounds__(NTHREADS, 1)
void cosine_wmma_pre_kernel(const unsigned short* __restrict__ x1hi,
                            const unsigned short* __restrict__ x1lo,
                            const unsigned short* __restrict__ x2hi,
                            const unsigned short* __restrict__ x2lo,
                            const float* __restrict__ n1, const float* __restrict__ n2,
                            float* __restrict__ out, int M, int N) {
  // 2 buffers x 4 planes (Ahi, Alo, Bhi, Blo) -> 80 KB of the WGP's 320 KB LDS
  __shared__ __align__(16) unsigned short smem[2][4][PLANE];

  const int tid  = threadIdx.x;
  const int lane = tid & 31;
  const int wave = tid >> 5;
  const int half = lane >> 4;
  const int l16  = lane & 15;
  const int wm   = wave & 3;      // 4 waves along M (32 rows each)
  const int wn   = wave >> 2;     // 2 waves along N (64 cols each)

  const int m0 = blockIdx.y * BM;
  const int n0 = blockIdx.x * BN;

  // Issue one chunk's tiles: 4 planes x 512 lane-jobs (row, 16B quarter),
  // 2 jobs per thread per plane -> 8 async b128 per wave per chunk.
  auto stage_async = [&](int buf, int kb) {
    const unsigned short* src[4] = {x2hi, x2lo, x1hi, x1lo};
    const int             rb[4]  = {m0, m0, n0, n0};
#pragma unroll
    for (int p = 0; p < 4; ++p) {
      unsigned short* lbase = &smem[buf][p][0];
#pragma unroll
      for (int i = 0; i < 2; ++i) {
        const int j = i * NTHREADS + tid;     // 0..511
        const int r = j >> 2;
        const int q = j & 3;
        async_copy16(src[p] + (size_t)(rb[p] + r) * D_DIM + kb + q * 8,
                     lbase + r * LDK + q * 8);
      }
    }
  };

  // 16-bit fragment layout (ISA 7.12.2): lane<16 -> K 0..7 & 16..23,
  // lane>=16 -> K 8..15 & 24..31; fragment row = lane & 15.
  auto load_frag = [&](const unsigned short* base, int row) -> v16bf {
    const unsigned short* p = base + row * LDK + half * 8;
    v8bf p0 = *(const v8bf*)(const void*)(p);
    v8bf p1 = *(const v8bf*)(const void*)(p + 16);
    return __builtin_shufflevector(p0, p1, 0, 1, 2, 3, 4, 5, 6, 7,
                                   8, 9, 10, 11, 12, 13, 14, 15);
  };

  v8f acc[2][4];
  {
    v8f z = {};
#pragma unroll
    for (int mi = 0; mi < 2; ++mi)
#pragma unroll
      for (int ni = 0; ni < 4; ++ni) acc[mi][ni] = z;
  }

  stage_async(0, 0);
  wait_async0();
  __syncthreads();

  int buf = 0;
  for (int t = 0; t < NCHUNK; ++t) {
    const bool more = (t + 1) < NCHUNK;
    if (more) stage_async(buf ^ 1, (t + 1) * KC);   // flies under the WMMAs

    const unsigned short* A_hi = &smem[buf][0][0];
    const unsigned short* A_lo = &smem[buf][1][0];
    const unsigned short* B_hi = &smem[buf][2][0];
    const unsigned short* B_lo = &smem[buf][3][0];

    v16bf bhf[4], blf[4];
#pragma unroll
    for (int ni = 0; ni < 4; ++ni) {
      const int br = wn * 64 + ni * 16 + l16;
      bhf[ni] = load_frag(B_hi, br);
      blf[ni] = load_frag(B_lo, br);
    }
#pragma unroll
    for (int mi = 0; mi < 2; ++mi) {
      const int ar = wm * 32 + mi * 16 + l16;
      v16bf ahf = load_frag(A_hi, ar);
      v16bf alf = load_frag(A_lo, ar);
#pragma unroll
      for (int ni = 0; ni < 4; ++ni) {
        // 3xBF16: hi*hi + hi*lo + lo*hi  (lo*lo ~ 2^-16, dropped)
        acc[mi][ni] = wmma_bf16(ahf, bhf[ni], acc[mi][ni]);
        acc[mi][ni] = wmma_bf16(ahf, blf[ni], acc[mi][ni]);
        acc[mi][ni] = wmma_bf16(alf, bhf[ni], acc[mi][ni]);
      }
    }

    if (more) {
      wait_async0();      // our copies into buf^1 landed
      __syncthreads();    // everyone's landed + everyone done reading buf
      buf ^= 1;
    }
  }

  // Epilogue: element j of v8f -> row (j + 8*half), col (lane & 15).
#pragma unroll
  for (int mi = 0; mi < 2; ++mi) {
#pragma unroll
    for (int ni = 0; ni < 4; ++ni) {
      const v8f a    = acc[mi][ni];
      const int row0 = m0 + wm * 32 + mi * 16 + half * 8;
      const int col  = n0 + wn * 64 + ni * 16 + l16;
      const float ncol = n1[col];
#pragma unroll
      for (int j = 0; j < 8; ++j) {
        const int row   = row0 + j;
        const float den = fmaxf(n2[row] * ncol, EPS_F);
        out[(size_t)row * N + col] = a[j] / den;
      }
    }
  }
}

// ---------------------------------------------------------------------------
// Fallback (workspace too small for planes): round-1 in-kernel conversion.
// ---------------------------------------------------------------------------
__global__ void row_norms_kernel(const float* __restrict__ x,
                                 float* __restrict__ nrm, int rows) {
  const int lane = threadIdx.x & 31;
  const int wave = threadIdx.x >> 5;
  const int row  = blockIdx.x * 8 + wave;
  if (row >= rows) return;
  const float* p = x + (size_t)row * D_DIM;
  float s = 0.f;
#pragma unroll
  for (int j = 0; j < D_DIM / 32; ++j) {
    const float v = p[lane + j * 32];
    s += v * v;
  }
#pragma unroll
  for (int off = 16; off > 0; off >>= 1) s += __shfl_xor(s, off, 32);
  if (lane == 0) nrm[row] = sqrtf(s);
}

__global__ __launch_bounds__(NTHREADS, 1)
void cosine_wmma_fallback_kernel(const float* __restrict__ x1, const float* __restrict__ x2,
                                 const float* __restrict__ n1, const float* __restrict__ n2,
                                 float* __restrict__ out, int M, int N) {
  __shared__ __align__(16) unsigned short smem[4 * BM * LDK];
  unsigned short* a_hi = smem;
  unsigned short* a_lo = smem + 1 * BM * LDK;
  unsigned short* b_hi = smem + 2 * BM * LDK;
  unsigned short* b_lo = smem + 3 * BM * LDK;

  const int tid  = threadIdx.x;
  const int lane = tid & 31;
  const int wave = tid >> 5;
  const int half = lane >> 4;
  const int l16  = lane & 15;
  const int wm   = wave & 3;
  const int wn   = wave >> 2;
  const int m0 = blockIdx.y * BM;
  const int n0 = blockIdx.x * BN;

  float4 aReg[4], bReg[4];
  auto issue_loads = [&](int kb) {
#pragma unroll
    for (int i = 0; i < 4; ++i) {
      const int f = i * NTHREADS + tid, r = f >> 3, c4 = f & 7;
      aReg[i] = *(const float4*)(x2 + (size_t)(m0 + r) * D_DIM + kb + c4 * 4);
      bReg[i] = *(const float4*)(x1 + (size_t)(n0 + r) * D_DIM + kb + c4 * 4);
    }
  };
  auto stage = [&]() {
#pragma unroll
    for (int i = 0; i < 4; ++i) {
      const int f = i * NTHREADS + tid, r = f >> 3, c4 = f & 7;
      const float av[4] = {aReg[i].x, aReg[i].y, aReg[i].z, aReg[i].w};
      const float bv[4] = {bReg[i].x, bReg[i].y, bReg[i].z, bReg[i].w};
      us4 ah, al, bh, bl;
#pragma unroll
      for (int c = 0; c < 4; ++c) {
        unsigned int h = f2bf_bits(av[c]);
        ah[c] = (unsigned short)h;
        al[c] = (unsigned short)f2bf_bits(av[c] - bfbits2f(h));
        h = f2bf_bits(bv[c]);
        bh[c] = (unsigned short)h;
        bl[c] = (unsigned short)f2bf_bits(bv[c] - bfbits2f(h));
      }
      const int o = r * LDK + c4 * 4;
      *(us4*)(a_hi + o) = ah; *(us4*)(a_lo + o) = al;
      *(us4*)(b_hi + o) = bh; *(us4*)(b_lo + o) = bl;
    }
  };
  auto load_frag = [&](const unsigned short* base, int row) -> v16bf {
    const unsigned short* p = base + row * LDK + half * 8;
    v8bf p0 = *(const v8bf*)(const void*)(p);
    v8bf p1 = *(const v8bf*)(const void*)(p + 16);
    return __builtin_shufflevector(p0, p1, 0, 1, 2, 3, 4, 5, 6, 7,
                                   8, 9, 10, 11, 12, 13, 14, 15);
  };

  v8f acc[2][4];
  {
    v8f z = {};
#pragma unroll
    for (int mi = 0; mi < 2; ++mi)
#pragma unroll
      for (int ni = 0; ni < 4; ++ni) acc[mi][ni] = z;
  }

  issue_loads(0);
  stage();
  __syncthreads();
  for (int kb = 0; kb < D_DIM; kb += KC) {
    const bool more = (kb + KC) < D_DIM;
    if (more) issue_loads(kb + KC);
    v16bf bhf[4], blf[4];
#pragma unroll
    for (int ni = 0; ni < 4; ++ni) {
      const int br = wn * 64 + ni * 16 + l16;
      bhf[ni] = load_frag(b_hi, br);
      blf[ni] = load_frag(b_lo, br);
    }
#pragma unroll
    for (int mi = 0; mi < 2; ++mi) {
      const int ar = wm * 32 + mi * 16 + l16;
      v16bf ahf = load_frag(a_hi, ar);
      v16bf alf = load_frag(a_lo, ar);
#pragma unroll
      for (int ni = 0; ni < 4; ++ni) {
        acc[mi][ni] = wmma_bf16(ahf, bhf[ni], acc[mi][ni]);
        acc[mi][ni] = wmma_bf16(ahf, blf[ni], acc[mi][ni]);
        acc[mi][ni] = wmma_bf16(alf, bhf[ni], acc[mi][ni]);
      }
    }
    __syncthreads();
    if (more) { stage(); __syncthreads(); }
  }
#pragma unroll
  for (int mi = 0; mi < 2; ++mi) {
#pragma unroll
    for (int ni = 0; ni < 4; ++ni) {
      const v8f a    = acc[mi][ni];
      const int row0 = m0 + wm * 32 + mi * 16 + half * 8;
      const int col  = n0 + wn * 64 + ni * 16 + l16;
      const float ncol = n1[col];
#pragma unroll
      for (int j = 0; j < 8; ++j) {
        const int row   = row0 + j;
        const float den = fmaxf(n2[row] * ncol, EPS_F);
        out[(size_t)row * N + col] = a[j] / den;
      }
    }
  }
}

extern "C" void kernel_launch(void* const* d_in, const int* in_sizes, int n_in,
                              void* d_out, int out_size, void* d_ws, size_t ws_size,
                              hipStream_t stream) {
  (void)n_in; (void)out_size;
  const float* x1 = (const float*)d_in[0];   // [N, 512]
  const float* x2 = (const float*)d_in[1];   // [M, 512]
  const int N = in_sizes[0] / D_DIM;
  const int M = in_sizes[1] / D_DIM;
  float* out = (float*)d_out;                // [M, N]

  float* n1 = (float*)d_ws;                  // N floats
  float* n2 = n1 + N;                        // M floats
  size_t norm_bytes = ((size_t)(M + N) * sizeof(float) + 255) & ~(size_t)255;
  unsigned short* x1hi = (unsigned short*)((char*)d_ws + norm_bytes);
  unsigned short* x1lo = x1hi + (size_t)N * D_DIM;
  unsigned short* x2hi = x1lo + (size_t)N * D_DIM;
  unsigned short* x2lo = x2hi + (size_t)M * D_DIM;
  const size_t need = norm_bytes + (size_t)2 * (M + N) * D_DIM * sizeof(unsigned short);

  dim3 grid(N / BN, M / BM);

  if (ws_size >= need) {
    convert_norms_kernel<<<(N + 7) / 8, 256, 0, stream>>>(x1, x1hi, x1lo, n1, N);
    convert_norms_kernel<<<(M + 7) / 8, 256, 0, stream>>>(x2, x2hi, x2lo, n2, M);
    cosine_wmma_pre_kernel<<<grid, NTHREADS, 0, stream>>>(x1hi, x1lo, x2hi, x2lo,
                                                          n1, n2, out, M, N);
  } else {
    row_norms_kernel<<<(N + 7) / 8, 256, 0, stream>>>(x1, n1, N);
    row_norms_kernel<<<(M + 7) / 8, 256, 0, stream>>>(x2, n2, M);
    cosine_wmma_fallback_kernel<<<grid, NTHREADS, 0, stream>>>(x1, x2, n1, n2,
                                                               out, M, N);
  }
}
